// QMogrifierLSTM_6004364280112
// MI455X (gfx1250) — matrix operand order repaired and checked by reading, not compiled, not run on hardware
//
#include <hip/hip_runtime.h>

// Mogrifier-style LSTM, persistent-grid WMMA kernel for gfx1250 (MI455X).
// B=128, S=1024, I=256, H=512.
// Grid: 128 WGs = 8 batch tiles (16 rows) x 16 column slices (32 h-cols each).
// Weights live in LDS as bf16 for the whole kernel; h/xt exchanged via L2.

#define B_SZ 128
#define S_SZ 1024
#define I_SZ 256
#define H_SZ 512
#define NWG  128
#define NTHR 256
#define WM_PITCH 528   // 513 padded to multiple of 8 elements (16B LDS loads)

typedef __bf16 bf16;
typedef __attribute__((ext_vector_type(16))) __bf16 v16bf;
typedef __attribute__((ext_vector_type(8)))  __bf16 v8bf;
typedef __attribute__((ext_vector_type(8)))  float  v8f;

__device__ __forceinline__ v8f wmma_bf16(v16bf a, v16bf b, v8f c) {
  // D = A(16x32 bf16) x B(32x16 bf16) + C(16x16 f32)
  return __builtin_amdgcn_wmma_f32_16x16x32_bf16(
      /*neg_a=*/false, a, /*neg_b=*/false, b,
      /*c_mod=*/(short)0, c, /*reuse_a=*/false, /*reuse_b=*/false);
}

// A-matrix tile (16 rows x K=32) from a row-major bf16 buffer.
// ISA layout: lanes 0-15 = rows, elems 0..7 -> K kc..kc+7, 8..15 -> K kc+16..23;
// lanes 16-31 shift K by +8.
__device__ __forceinline__ v16bf load_a_tile(const bf16* __restrict__ base,
                                             int stride, int kc, int lane) {
  const int m  = lane & 15;
  const int hf = lane >> 4;
  const bf16* p = base + m * stride + kc + 8 * hf;
  v8bf lo = *(const v8bf*)p;          // 16B global/ds load
  v8bf hi = *(const v8bf*)(p + 16);   // 16B global/ds load
  return __builtin_shufflevector(lo, hi, 0,1,2,3,4,5,6,7,8,9,10,11,12,13,14,15);
}

// B-matrix tile (K=32 x 16 cols) taken as W^T, i.e. from row-major W(out,in):
// lane n (0-15) = output column = W row; elems 0..7 -> K kc..kc+7, 8..15 -> K kc+16..23;
// lanes 16-31 shift K by +8 (same K order as load_a_tile).
__device__ __forceinline__ v16bf load_b_tile(const bf16* tile, int stride,
                                             int kc, int lane) {
  const int n  = lane & 15;
  const int kh = lane >> 4;
  const bf16* p = tile + n * stride + kc + 8 * kh;
  v8bf lo = *(const v8bf*)p;
  v8bf hi = *(const v8bf*)(p + 16);
  return __builtin_shufflevector(lo, hi, 0,1,2,3,4,5,6,7,8,9,10,11,12,13,14,15);
}

__device__ __forceinline__ float sigm(float x) { return 1.0f / (1.0f + __expf(-x)); }

// Device-wide sense-free barrier: monotonically increasing epoch target.
__device__ __forceinline__ void gbar(unsigned* cnt, unsigned target) {
  __threadfence();
  __syncthreads();
  if (threadIdx.x == 0) {
    __hip_atomic_fetch_add(cnt, 1u, __ATOMIC_RELEASE, __HIP_MEMORY_SCOPE_AGENT);
    while (__hip_atomic_load(cnt, __ATOMIC_ACQUIRE, __HIP_MEMORY_SCOPE_AGENT) < target)
      __builtin_amdgcn_s_sleep(2);
  }
  __syncthreads();
  __threadfence();
}

extern __shared__ __align__(32) char smem_raw[];

__global__ void __launch_bounds__(NTHR, 1)
mog_lstm_wmma(const float* __restrict__ x, const float* __restrict__ dlt,
              const float* __restrict__ W_m, const float* __restrict__ b_m,
              const float* __restrict__ W_ih, const float* __restrict__ b_ih,
              const float* __restrict__ W_hh, const float* __restrict__ b_hh,
              float* __restrict__ out,
              unsigned* __restrict__ barrier_cnt,
              bf16* __restrict__ h_bf, bf16* __restrict__ xt_bf)
{
  // LDS carve: persistent weight slices + staging
  bf16*  sW_hh = (bf16*)smem_raw;                 // 128 x 512   (128 KB)
  bf16*  sW_ih = sW_hh + 128 * H_SZ;              // 128 x 256   ( 64 KB)
  bf16*  sW_m  = sW_ih + 128 * I_SZ;              // 16 x 528    (~17 KB)
  float* sBias = (float*)(sW_m + 16 * WM_PITCH);  // 128
  float* sGate = sBias + 128;                     // 16 x 128    (  8 KB)
  float* sC    = sGate + 16 * 128;                // 16 x 32 c-state

  const int tid  = threadIdx.x;
  const int lane = tid & 31;          // wave32
  const int wave = tid >> 5;          // 0..7, one 16x16 gate tile each
  const int kslc = blockIdx.x & 15;   // h-column slice: cols [32*kslc, +32)
  const int bt   = blockIdx.x >> 4;   // batch tile: rows [16*bt, +16)

  // Gate tile -> global weight-row mapping.
  // wave w: gate g=w>>1, sub=w&1 -> W row base g*512 + 32*kslc + 16*sub.
  // --- one-time: stage weight slices into LDS as bf16 ------------------
  for (int idx = tid; idx < 128 * H_SZ; idx += NTHR) {
    const int r = idx >> 9, c = idx & (H_SZ - 1);
    const int w = r >> 4, n = r & 15;
    const int grow = (w >> 1) * H_SZ + 32 * kslc + 16 * (w & 1) + n;
    sW_hh[idx] = (bf16)W_hh[grow * H_SZ + c];
  }
  for (int idx = tid; idx < 128 * I_SZ; idx += NTHR) {
    const int r = idx >> 8, c = idx & (I_SZ - 1);
    const int w = r >> 4, n = r & 15;
    const int grow = (w >> 1) * H_SZ + 32 * kslc + 16 * (w & 1) + n;
    sW_ih[idx] = (bf16)W_ih[grow * I_SZ + c];
  }
  for (int idx = tid; idx < 16 * WM_PITCH; idx += NTHR) {
    const int r = idx / WM_PITCH, c = idx % WM_PITCH;
    sW_m[idx] = (c < H_SZ + 1) ? (bf16)W_m[(16 * kslc + r) * (H_SZ + 1) + c]
                               : (bf16)0.0f;
  }
  for (int c = tid; c < 128; c += NTHR) {
    const int w = c >> 4, n = c & 15;
    const int grow = (w >> 1) * H_SZ + 32 * kslc + 16 * (w & 1) + n;
    sBias[c] = b_ih[grow] + b_hh[grow];
  }
  for (int i = tid; i < 16 * 32; i += NTHR) sC[i] = 0.0f;
  for (int i = tid; i < 16 * 32; i += NTHR) {       // h0 = 0 (our slice)
    const int b = i >> 5, j = i & 31;
    h_bf[(bt * 16 + b) * H_SZ + 32 * kslc + j] = (bf16)0.0f;
  }

  unsigned ep = 0;
  gbar(barrier_cnt, ++ep * NWG);   // weights + h0 visible everywhere

  const bf16* hA  = h_bf  + (size_t)bt * 16 * H_SZ;   // A: 16 x 512 bf16
  const bf16* xtA = xt_bf + (size_t)bt * 16 * I_SZ;   // A: 16 x 256 bf16
  const bf16* wBh = sW_hh + wave * 16 * H_SZ;         // B tiles in LDS
  const bf16* wBi = sW_ih + wave * 16 * I_SZ;
  const float biasv = sBias[wave * 16 + (lane & 15)];

  for (int t = 0; t < S_SZ; ++t) {
    // ---- gates, W_hh half (h already valid) ------------------------------
    v8f acc;
    #pragma unroll
    for (int r = 0; r < 8; ++r) acc[r] = biasv;
    #pragma unroll 4
    for (int kc = 0; kc < H_SZ; kc += 32)
      acc = wmma_bf16(load_a_tile(hA, H_SZ, kc, lane),
                      load_b_tile(wBh, H_SZ, kc, lane), acc);

    // ---- modulation slice: m = sigmoid([h,d] @ W_m^T + b_m), xt = m*x ----
    if (wave == 0) {   // wave-uniform branch: EXEC all-ones inside (WMMA req)
      v8f mac = {};
      #pragma unroll 4
      for (int kc = 0; kc < H_SZ; kc += 32)
        mac = wmma_bf16(load_a_tile(hA, H_SZ, kc, lane),
                        load_b_tile(sW_m, WM_PITCH, kc, lane), mac);
      const int n = lane & 15, hf = lane >> 4;
      const float wd = (float)sW_m[n * WM_PITCH + H_SZ];  // delta column K=512
      const float bm = b_m[16 * kslc + n];
      #pragma unroll
      for (int r = 0; r < 8; ++r) {
        const int b = r + 8 * hf;                     // C-layout row
        const float dv = dlt[(bt * 16 + b) * S_SZ + t];
        const float mv = sigm(mac[r] + bm + wd * dv);
        const float xv = x[((size_t)(bt * 16 + b) * S_SZ + t) * I_SZ
                           + 16 * kslc + n];
        xt_bf[(bt * 16 + b) * I_SZ + 16 * kslc + n] = (bf16)(mv * xv);
      }
    }

    gbar(barrier_cnt, ++ep * NWG);   // all xt slices visible

    // ---- gates, W_ih half ------------------------------------------------
    #pragma unroll 4
    for (int kc = 0; kc < I_SZ; kc += 32)
      acc = wmma_bf16(load_a_tile(xtA, I_SZ, kc, lane),
                      load_b_tile(wBi, I_SZ, kc, lane), acc);

    // ---- stage accumulators, recombine i/f/g/o per h-column --------------
    {
      const int n = lane & 15, hf = lane >> 4;
      #pragma unroll
      for (int r = 0; r < 8; ++r)
        sGate[(r + 8 * hf) * 128 + wave * 16 + n] = acc[r];
    }
    __syncthreads();

    for (int e = tid; e < 16 * 32; e += NTHR) {   // 2 elems/thread
      const int b = e >> 5, j = e & 31;           // col c = g*32 + j
      const float gi = sGate[b * 128 +       j];
      const float gf = sGate[b * 128 +  32 + j];
      const float gg = sGate[b * 128 +  64 + j];
      const float go = sGate[b * 128 +  96 + j];
      const float cn = sigm(gf) * sC[e] + sigm(gi) * tanhf(gg);
      const float hn = sigm(go) * tanhf(cn);
      sC[e] = cn;
      h_bf[(bt * 16 + b) * H_SZ + 32 * kslc + j] = (bf16)hn;
      out[((size_t)(bt * 16 + b) * S_SZ + t) * H_SZ + 32 * kslc + j] = hn;
      if (t == S_SZ - 1)
        out[(size_t)B_SZ * S_SZ * H_SZ + (bt * 16 + b) * H_SZ
            + 32 * kslc + j] = hn;
    }

    gbar(barrier_cnt, ++ep * NWG);   // h_{t+1} visible everywhere
  }
}

extern "C" void kernel_launch(void* const* d_in, const int* in_sizes, int n_in,
                              void* d_out, int out_size, void* d_ws, size_t ws_size,
                              hipStream_t stream) {
  (void)in_sizes; (void)n_in; (void)out_size; (void)ws_size;
  const float* x    = (const float*)d_in[0];
  const float* dlt  = (const float*)d_in[1];
  const float* W_m  = (const float*)d_in[2];
  const float* b_m  = (const float*)d_in[3];
  const float* W_ih = (const float*)d_in[4];
  const float* b_ih = (const float*)d_in[5];
  const float* W_hh = (const float*)d_in[6];
  const float* b_hh = (const float*)d_in[7];
  float* out = (float*)d_out;

  unsigned* cnt = (unsigned*)d_ws;                       // barrier counter
  bf16* h_bf  = (bf16*)((char*)d_ws + 256);              // 128x512 bf16
  bf16* xt_bf = h_bf + (size_t)B_SZ * H_SZ;              // 128x256 bf16

  hipMemsetAsync(d_ws, 0, 256, stream);                  // reset barrier state

  const size_t lds_bytes =
      (size_t)128 * H_SZ * 2 +     // sW_hh
      (size_t)128 * I_SZ * 2 +     // sW_ih
      (size_t)16 * WM_PITCH * 2 +  // sW_m
      128 * 4 +                    // sBias
      16 * 128 * 4 +               // sGate
      16 * 32 * 4;                 // sC   => 224,256 B (< 320 KB WGP LDS)

  mog_lstm_wmma<<<dim3(NWG), dim3(NTHR), lds_bytes, stream>>>(
      x, dlt, W_m, b_m, W_ih, b_ih, W_hh, b_hh, out, cnt, h_bf, xt_bf);
}
